// RNNTDecoder_33990371181317
// MI455X (gfx1250) — matrix operand order repaired and checked by reading, not compiled, not run on hardware
//
#include <hip/hip_runtime.h>
#include <hip/hip_bf16.h>
#include <math.h>

// ---------------------------------------------------------------------------
// RNN-T decoder for MI455X (gfx1250, wave32, WMMA).
// Dominant cost: joint GEMM [65536 x 1025] = relu(enc_p (+) pred_p) @ Wo,
// done in bf16 WMMA (f32 accumulate). All GEMM operands are L2-resident
// (enc_p 2.6MB + pred_p 0.65MB + Wot 1.3MB << 192MB L2); the 268MB
// non-temporal output write is the HBM roofline floor (~11.5us @ 23.3TB/s).
// ---------------------------------------------------------------------------

#define B_  4
#define T_  256
#define U_  64
#define DE_ 512
#define H_  640
#define G4_ (4 * H_)      // 2560
#define V_  1024
#define VO_ (V_ + 1)      // 1025
#define MTOT (B_ * T_ * U_)  // 65536

typedef __attribute__((ext_vector_type(16))) __bf16 v16bf;
typedef __attribute__((ext_vector_type(8)))  float  v8f;

// ---------------------------------------------------------------------------
// 1) Embedding gather: x[b,u,:] = embed[targets[b,u], :]
// ---------------------------------------------------------------------------
__global__ void embed_gather(const int* __restrict__ targets,
                             const float* __restrict__ embed,
                             float* __restrict__ x) {
  int m = blockIdx.x * blockDim.x + threadIdx.x;      // < B*U*H
  if (m >= B_ * U_ * H_) return;
  int h  = m % H_;
  int bu = m / H_;
  int tg = targets[bu];
  x[m] = embed[(size_t)tg * H_ + h];
}

// ---------------------------------------------------------------------------
// 2) Generic fp32 tiled SGEMM with bias: C[M,N] = A[M,K] @ W[K,N] + bias[N]
//    (all M,N,K multiples of 16 in this workload). Used for the small
//    projections; kept fp32 since xz0 feeds the LSTM recurrence.
// ---------------------------------------------------------------------------
__global__ __launch_bounds__(256) void sgemm_bias(
    const float* __restrict__ A, const float* __restrict__ W,
    const float* __restrict__ bias, float* __restrict__ C,
    int M, int N, int K) {
  __shared__ float As[16][16];
  __shared__ float Ws[16][17];
  int tx = threadIdx.x, ty = threadIdx.y;
  int row = blockIdx.y * 16 + ty;
  int col = blockIdx.x * 16 + tx;
  float acc = 0.f;
  for (int k0 = 0; k0 < K; k0 += 16) {
    As[ty][tx] = A[(size_t)row * K + k0 + tx];
    Ws[ty][tx] = W[(size_t)(k0 + ty) * N + col];
    __syncthreads();
#pragma unroll
    for (int i = 0; i < 16; ++i) acc += As[ty][i] * Ws[i][tx];
    __syncthreads();
  }
  C[(size_t)row * N + col] = acc + bias[col];
}

// ---------------------------------------------------------------------------
// 3) Pack Wo[H, V+1] -> column-major bf16 Wot[V][H] (WMMA B operand feed)
//    + last (blank) column kept fp32 contiguous.
// ---------------------------------------------------------------------------
__global__ void pack_wo(const float* __restrict__ Wo,
                        __bf16* __restrict__ Wot,
                        float* __restrict__ woLast) {
  int t = blockIdx.x * blockDim.x + threadIdx.x;      // < H*V
  if (t >= H_ * V_) return;
  int n = t / H_;
  int k = t % H_;
  Wot[(size_t)n * H_ + k] = (__bf16)Wo[(size_t)k * VO_ + n];
  if (t < H_) woLast[t] = Wo[(size_t)t * VO_ + V_];
}

// ---------------------------------------------------------------------------
// 4) LSTM step (fp32, sequential over U). One launch handles layer-0 step k
//    (blocks 0..9) and layer-1 step k-1 (blocks 10..19) -> 65 launches total.
//    Layer-1 folds its input projection (h0 @ Wx1) into the step so no
//    separate batched xz1 GEMM is needed.
// ---------------------------------------------------------------------------
__device__ __forceinline__ float sigmoidf_(float x) {
  return 1.f / (1.f + __expf(-x));
}

__global__ __launch_bounds__(256) void lstm_step(
    int k,
    const float* __restrict__ xz0,   // [B,U,4H] (bias b0 already added)
    const float* __restrict__ Wh0,   // [H,4H]
    const float* __restrict__ Wx1,   // [H,4H]
    const float* __restrict__ Wh1,   // [H,4H]
    const float* __restrict__ b1,    // [4H]
    float* __restrict__ h0seq,       // [B,U,H]
    float* __restrict__ h1seq,       // [B,U,H]  (= pred)
    float* __restrict__ c0,          // [B,H]
    float* __restrict__ c1) {        // [B,H]
  int layer = blockIdx.x / 10;
  int idx = (blockIdx.x % 10) * 256 + threadIdx.x;    // < B*H = 2560
  int b = idx / H_;
  int j = idx % H_;

  if (layer == 0) {
    if (k >= U_) return;
    float zi = xz0[(size_t)(b * U_ + k) * G4_ + j];
    float zf = xz0[(size_t)(b * U_ + k) * G4_ + H_ + j];
    float zg = xz0[(size_t)(b * U_ + k) * G4_ + 2 * H_ + j];
    float zo = xz0[(size_t)(b * U_ + k) * G4_ + 3 * H_ + j];
    if (k > 0) {
      const float* hp = h0seq + (size_t)(b * U_ + (k - 1)) * H_;
      for (int h = 0; h < H_; ++h) {
        float hv = hp[h];
        const float* wr = Wh0 + (size_t)h * G4_ + j;
        zi = fmaf(hv, wr[0],      zi);
        zf = fmaf(hv, wr[H_],     zf);
        zg = fmaf(hv, wr[2 * H_], zg);
        zo = fmaf(hv, wr[3 * H_], zo);
      }
    }
    float cp = (k > 0) ? c0[idx] : 0.f;
    float c = sigmoidf_(zf) * cp + sigmoidf_(zi) * tanhf(zg);
    c0[idx] = c;
    h0seq[(size_t)(b * U_ + k) * H_ + j] = sigmoidf_(zo) * tanhf(c);
  } else {
    if (k == 0) return;
    int kk = k - 1;                                   // layer-1 step index
    float zi = b1[j];
    float zf = b1[H_ + j];
    float zg = b1[2 * H_ + j];
    float zo = b1[3 * H_ + j];
    const float* x1 = h0seq + (size_t)(b * U_ + kk) * H_;
    const float* hp = (kk > 0) ? (h1seq + (size_t)(b * U_ + kk - 1) * H_)
                               : nullptr;
    for (int h = 0; h < H_; ++h) {
      float xv = x1[h];
      const float* wx = Wx1 + (size_t)h * G4_ + j;
      zi = fmaf(xv, wx[0],      zi);
      zf = fmaf(xv, wx[H_],     zf);
      zg = fmaf(xv, wx[2 * H_], zg);
      zo = fmaf(xv, wx[3 * H_], zo);
      if (hp) {
        float hv = hp[h];
        const float* wh = Wh1 + (size_t)h * G4_ + j;
        zi = fmaf(hv, wh[0],      zi);
        zf = fmaf(hv, wh[H_],     zf);
        zg = fmaf(hv, wh[2 * H_], zg);
        zo = fmaf(hv, wh[3 * H_], zo);
      }
    }
    float cp = (kk > 0) ? c1[idx] : 0.f;
    float c = sigmoidf_(zf) * cp + sigmoidf_(zi) * tanhf(zg);
    c1[idx] = c;
    h1seq[(size_t)(b * U_ + kk) * H_ + j] = sigmoidf_(zo) * tanhf(c);
  }
}

// ---------------------------------------------------------------------------
// 5) Joint GEMM: out[m, n] = relu(enc_p[b,t,:] + pred_p[b,u,:]) . Wot[n,:]
//    bf16 WMMA, f32 accumulate. Grid (512, 8) x 256 threads (8 waves).
//    Wave w: M strip = blockIdx.x*128 + w*16, 8 accumulator tiles over 128 N
//    (A-fragment build amortized over 8 WMMAs -> 3 b128 loads per WMMA).
//    A fragment built in registers, contiguous K per lane half:
//      lanes 0-15 : M=lane,    K = [kk..kk+15]
//      lanes 16-31: M=lane-16, K = [kk+16..kk+31]
//    B fragment: same K order as A = 16 contiguous K halves per lane from
//    column-major Wot (lanes 0-15: K lo 16, lanes 16-31: K hi 16).
//    Output stores are non-temporal: 268MB streaming write, never re-read,
//    keeps L2 clean for the resident GEMM operands.
// ---------------------------------------------------------------------------
#define NT_ 8   // N tiles (of 16) per wave

__global__ __launch_bounds__(256) void joint_wmma(
    const float* __restrict__ enc_p,   // [B,T,H]
    const float* __restrict__ pred_p,  // [B,U,H]
    const __bf16* __restrict__ Wot,    // [V,H] col-major (n-major)
    const float* __restrict__ bo,      // [V+1]
    float* __restrict__ out) {         // [B,T,U,V+1]
  int lane = threadIdx.x & 31;
  int wave = threadIdx.x >> 5;
  int m_base = blockIdx.x * 128 + wave * 16;
  int n_base = blockIdx.y * (16 * NT_);

  // A-row for this lane (16 | U so the strip shares one (b,t) enc row)
  int mA = m_base + (lane & 15);
  int b = mA >> 14;              // / (T*U)
  int t = (mA >> 6) & (T_ - 1);
  int u = mA & (U_ - 1);
  const float* encR  = enc_p  + (size_t)(b * T_ + t) * H_;
  const float* predR = pred_p + (size_t)(b * U_ + u) * H_;
  int hi = (lane >> 4) & 1;
  int kShiftA = hi ? 16 : 0;
  const __bf16* bCol = Wot + (size_t)(n_base + (lane & 15)) * H_ + (hi ? 16 : 0);

  v8f acc[NT_];
#pragma unroll
  for (int j = 0; j < NT_; ++j) acc[j] = (v8f){};

  for (int kk = 0; kk < H_; kk += 32) {
    // Build bf16 A fragment from relu(enc + pred)
    v16bf a;
    int k0 = kk + kShiftA;
#pragma unroll
    for (int i = 0; i < 8; ++i) {
      float v0 = encR[k0 + i]      + predR[k0 + i];
      float v1 = encR[k0 + 8 + i] + predR[k0 + 8 + i];
      v0 = v0 > 0.f ? v0 : 0.f;
      v1 = v1 > 0.f ? v1 : 0.f;
      a[i]     = (__bf16)v0;
      a[8 + i] = (__bf16)v1;
    }
    // 32-byte aligned vector loads of B fragments (L2-resident)
#pragma unroll
    for (int j = 0; j < NT_; ++j) {
      v16bf bb = *(const v16bf*)(bCol + (size_t)(16 * j) * H_ + kk);
      acc[j] = __builtin_amdgcn_wmma_f32_16x16x32_bf16(
          false, a, false, bb, (short)0, acc[j], false, false);
    }
  }

  // C/D layout: VGPR r -> M = r + 8*hi ; N = lane & 15
  int mOut = m_base + (hi << 3);
  int nn = n_base + (lane & 15);
  float bb[NT_];
#pragma unroll
  for (int j = 0; j < NT_; ++j) bb[j] = bo[nn + 16 * j];
#pragma unroll
  for (int r = 0; r < 8; ++r) {
    float* op = out + (size_t)(mOut + r) * VO_ + nn;
#pragma unroll
    for (int j = 0; j < NT_; ++j) {
      __builtin_nontemporal_store(acc[j][r] + bb[j], op + 16 * j);
    }
  }
}

// ---------------------------------------------------------------------------
// 6) Blank (last) vocab column, fp32: out[m, 1024]
// ---------------------------------------------------------------------------
__global__ void blank_col(const float* __restrict__ enc_p,
                          const float* __restrict__ pred_p,
                          const float* __restrict__ woLast,
                          const float* __restrict__ bo,
                          float* __restrict__ out) {
  int m = blockIdx.x * blockDim.x + threadIdx.x;      // < MTOT
  if (m >= MTOT) return;
  int b = m >> 14;
  int t = (m >> 6) & (T_ - 1);
  int u = m & (U_ - 1);
  const float* e = enc_p  + (size_t)(b * T_ + t) * H_;
  const float* p = pred_p + (size_t)(b * U_ + u) * H_;
  float acc = bo[V_];
  for (int k = 0; k < H_; ++k) {
    float v = e[k] + p[k];
    v = v > 0.f ? v : 0.f;
    acc = fmaf(v, woLast[k], acc);
  }
  __builtin_nontemporal_store(acc, out + (size_t)m * VO_ + V_);
}

// ---------------------------------------------------------------------------
// Host launcher
// ---------------------------------------------------------------------------
static inline size_t align256(size_t x) { return (x + 255) & ~(size_t)255; }

extern "C" void kernel_launch(void* const* d_in, const int* in_sizes, int n_in,
                              void* d_out, int out_size, void* d_ws, size_t ws_size,
                              hipStream_t stream) {
  const float* encoder_out = (const float*)d_in[0];
  const int*   targets     = (const int*)  d_in[1];
  const float* embed       = (const float*)d_in[2];
  const float* Wx0         = (const float*)d_in[3];
  const float* Wh0         = (const float*)d_in[4];
  const float* b0          = (const float*)d_in[5];
  const float* Wx1         = (const float*)d_in[6];
  const float* Wh1         = (const float*)d_in[7];
  const float* b1          = (const float*)d_in[8];
  const float* We          = (const float*)d_in[9];
  const float* be          = (const float*)d_in[10];
  const float* Wp          = (const float*)d_in[11];
  const float* bp          = (const float*)d_in[12];
  const float* Wo          = (const float*)d_in[13];
  const float* bo          = (const float*)d_in[14];
  float* out = (float*)d_out;

  // Workspace bump allocator
  char* w = (char*)d_ws;
  size_t off = 0;
  auto alloc = [&](size_t bytes) {
    void* p = w + off;
    off += align256(bytes);
    return p;
  };
  float*  x_emb  = (float*) alloc((size_t)B_ * U_ * H_  * 4);   // [B,U,H]
  float*  xz0    = (float*) alloc((size_t)B_ * U_ * G4_ * 4);   // [B,U,4H]
  float*  h0seq  = (float*) alloc((size_t)B_ * U_ * H_  * 4);
  float*  h1seq  = (float*) alloc((size_t)B_ * U_ * H_  * 4);   // = pred
  float*  c0     = (float*) alloc((size_t)B_ * H_ * 4);
  float*  c1     = (float*) alloc((size_t)B_ * H_ * 4);
  float*  enc_p  = (float*) alloc((size_t)B_ * T_ * H_ * 4);
  float*  pred_p = (float*) alloc((size_t)B_ * U_ * H_ * 4);
  __bf16* Wot    = (__bf16*)alloc((size_t)V_ * H_ * 2);
  float*  woLast = (float*) alloc((size_t)H_ * 4);
  (void)ws_size; (void)in_sizes; (void)n_in; (void)out_size;

  // 1) embedding
  embed_gather<<<(B_ * U_ * H_ + 255) / 256, 256, 0, stream>>>(targets, embed, x_emb);

  // 2) xz0 = x @ Wx0 + b0   [256 x 2560, K=640]
  sgemm_bias<<<dim3(G4_ / 16, (B_ * U_) / 16), dim3(16, 16), 0, stream>>>(
      x_emb, Wx0, b0, xz0, B_ * U_, G4_, H_);

  // 3) enc_p = encoder_out @ We + be   [1024 x 640, K=512]
  sgemm_bias<<<dim3(H_ / 16, (B_ * T_) / 16), dim3(16, 16), 0, stream>>>(
      encoder_out, We, be, enc_p, B_ * T_, H_, DE_);

  // 4) pack Wo -> bf16 col-major + blank column
  pack_wo<<<(H_ * V_ + 255) / 256, 256, 0, stream>>>(Wo, Wot, woLast);

  // 5) LSTM: 65 pipelined launches (layer0 step k || layer1 step k-1)
  for (int k = 0; k <= U_; ++k) {
    lstm_step<<<20, 256, 0, stream>>>(k, xz0, Wh0, Wx1, Wh1, b1,
                                      h0seq, h1seq, c0, c1);
  }

  // 6) pred_p = pred @ Wp + bp   [256 x 640, K=640]
  sgemm_bias<<<dim3(H_ / 16, (B_ * U_) / 16), dim3(16, 16), 0, stream>>>(
      h1seq, Wp, bp, pred_p, B_ * U_, H_, H_);

  // 7) joint GEMM, bf16 WMMA: columns 0..1023
  joint_wmma<<<dim3(MTOT / 128, V_ / (16 * NT_)), 256, 0, stream>>>(
      enc_p, pred_p, Wot, bo, out);

  // 8) blank column 1024
  blank_col<<<MTOT / 256, 256, 0, stream>>>(enc_p, pred_p, woLast, bo, out);
}
